// CrossMambaBlock_7834020347952
// MI455X (gfx1250) — matrix operand order repaired and checked by reading, not compiled, not run on hardware
//
#include <hip/hip_runtime.h>
#include <hip/hip_bf16.h>

// ---------------- problem constants ----------------
constexpr int BN   = 2;            // batch
constexpr int LL   = 16384;        // sequence length (128*128)
constexpr int DM   = 256;          // model dim == inner dim d
constexpr int NS   = 8;            // D_STATE
constexpr int DTR  = 16;           // DT_RANK
constexpr int KC   = 4;            // D_CONV
constexpr int BL   = BN * LL;                       // 32768 rows
constexpr long long BLD = (long long)BL * DM;       // 8388608 elems
constexpr int CH   = 128;          // scan chunk length
constexpr int NC   = LL / CH;      // 128 chunks

// ---------------- types ----------------
typedef __attribute__((ext_vector_type(16))) __bf16 v16bf;
typedef __attribute__((ext_vector_type(8)))  float  v8f;
typedef __attribute__((ext_vector_type(4)))  float  f4raw;
union Frag { v16bf v; f4raw f4[2]; };

// ---------------- helpers ----------------
__device__ __forceinline__ float sigmoidf_(float x) { return 1.0f / (1.0f + __expf(-x)); }
__device__ __forceinline__ float siluf_(float x)    { return x * sigmoidf_(x); }
__device__ __forceinline__ float softplusf_(float x){ return (x > 20.0f) ? x : log1pf(__expf(x)); }
__device__ __forceinline__ __bf16 f2bf(float f) {
  unsigned u = __float_as_uint(f);
  u += 0x7fffu + ((u >> 16) & 1u);            // round-to-nearest-even
  unsigned short s = (unsigned short)(u >> 16);
  __bf16 r; __builtin_memcpy(&r, &s, 2); return r;
}

// ---------------- f32 -> bf16 weight conversion ----------------
__global__ void cvt_f32_bf16(const float* __restrict__ in, __bf16* __restrict__ out, int n) {
  int i = blockIdx.x * blockDim.x + threadIdx.x;
  if (i < n) out[i] = f2bf(in[i]);
}

// ---------------- layernorm (per 256-wide row) -> bf16 ----------------
__global__ void __launch_bounds__(256)
ln_to_bf16(const float* __restrict__ x, const float* __restrict__ w,
           const float* __restrict__ b, __bf16* __restrict__ out) {
  const int row = blockIdx.x;
  const int c = threadIdx.x;
  const float v = x[(size_t)row * DM + c];
  __shared__ float red[DM];
  red[c] = v; __syncthreads();
  for (int s = 128; s > 0; s >>= 1) { if (c < s) red[c] += red[c + s]; __syncthreads(); }
  const float mu = red[0] * (1.0f / DM);
  __syncthreads();
  const float dv = v - mu;
  red[c] = dv * dv; __syncthreads();
  for (int s = 128; s > 0; s >>= 1) { if (c < s) red[c] += red[c + s]; __syncthreads(); }
  const float var = red[0] * (1.0f / DM);
  const float y = dv * rsqrtf(var + 1e-5f) * w[c] + b[c];
  out[(size_t)row * DM + c] = f2bf(y);
}

// ---------------- bf16 WMMA GEMM: out[m,n] = sum_k A[m,k]*W[n,k] ----------------
// mode 0: out0[m*N + n] = acc
// mode 1: split columns at 256 -> out0 (x), out1 (z), both stride 256
// mode 2: out0[m*256+n] = acc + skip[m*256+n]
__global__ void __launch_bounds__(128)
gemm_bf16_wmma(const __bf16* __restrict__ A, const __bf16* __restrict__ W,
               int M, int N, int K, int mode,
               float* __restrict__ out0, float* __restrict__ out1,
               const float* __restrict__ skip) {
  const int lane = threadIdx.x & 31;
  const int wave = threadIdx.x >> 5;
  const int ntiles = N >> 4;
  const int mtiles = M >> 4;
  const int tile = blockIdx.x * 4 + wave;
  if (tile >= mtiles * ntiles) return;
  const int mt = tile / ntiles;
  const int nt = tile - mt * ntiles;
  const int lg = lane >> 4;      // half-K group per ISA A/B fragment layout
  const int lr = lane & 15;
  const size_t arow = (size_t)(mt * 16 + lr) * K;
  const size_t wrow = (size_t)(nt * 16 + lr) * K;
  v8f acc = {0.f, 0.f, 0.f, 0.f, 0.f, 0.f, 0.f, 0.f};
  for (int kb = 0; kb < K; kb += 32) {
    Frag a, b;
    // A 16x32 bf16 fragment: lanes 0-15 hold K[kb..kb+8)+K[kb+16..kb+24),
    //                        lanes 16-31 hold K[kb+8..kb+16)+K[kb+24..kb+32)
    a.f4[0] = *reinterpret_cast<const f4raw*>(A + arow + kb + lg * 8);
    a.f4[1] = *reinterpret_cast<const f4raw*>(A + arow + kb + 16 + lg * 8);
    // B 32x16 fragment: column n = lr, same K split as A
    b.f4[0] = *reinterpret_cast<const f4raw*>(W + wrow + kb + lg * 8);
    b.f4[1] = *reinterpret_cast<const f4raw*>(W + wrow + kb + 16 + lg * 8);
    acc = __builtin_amdgcn_wmma_f32_16x16x32_bf16(false, a.v, false, b.v,
                                                  (short)0, acc, false, false);
  }
  const int n = nt * 16 + lr;
#pragma unroll
  for (int r = 0; r < 8; ++r) {
    const int m = mt * 16 + lg * 8 + r;      // C/D layout: VGPR r -> row r (+8 for hi lanes)
    const float v = acc[r];
    if (mode == 1) {
      if (n < 256) out0[(size_t)m * 256 + n] = v;
      else         out1[(size_t)m * 256 + (n - 256)] = v;
    } else if (mode == 2) {
      out0[(size_t)m * 256 + n] = v + skip[(size_t)m * 256 + n];
    } else {
      out0[(size_t)m * N + n] = v;
    }
  }
}

// ---------------- causal depthwise conv (K=4) + SiLU; rev => input read flipped --------
__global__ void __launch_bounds__(256)
conv_silu(const float* __restrict__ x, const float* __restrict__ w,
          const float* __restrict__ bias, float* __restrict__ out, int rev) {
  const long long idx = (long long)blockIdx.x * blockDim.x + threadIdx.x;
  if (idx >= BLD) return;
  const int c = (int)(idx & (DM - 1));
  const long long bt = idx >> 8;
  const int t = (int)(bt % LL);
  const int b = (int)(bt / LL);
  float acc = bias[c];
#pragma unroll
  for (int k = 0; k < KC; ++k) {
    const int tt = t - (KC - 1) + k;
    if (tt < 0) continue;
    const int st = rev ? (LL - 1 - tt) : tt;
    acc += w[c * KC + k] * x[((size_t)b * LL + st) * DM + c];
  }
  out[idx] = siluf_(acc);
}

// -------- fused: ec = silu(conv(xe)) then dbl[0:32] = ec @ xproj_w.T (per position) -----
__global__ void __launch_bounds__(256)
conve_xproj(const float* __restrict__ xe, const float* __restrict__ cw,
            const float* __restrict__ cb, const float* __restrict__ xpw,
            float* __restrict__ dbl, int rev) {
  const int pos = blockIdx.x;                 // b*L + t (scan-time order)
  const int t = pos & (LL - 1);
  const int b = pos >> 14;
  const int c = threadIdx.x;
  float acc = cb[c];
#pragma unroll
  for (int k = 0; k < KC; ++k) {
    const int tt = t - (KC - 1) + k;
    if (tt < 0) continue;
    const int st = rev ? (LL - 1 - tt) : tt;
    acc += cw[c * KC + k] * xe[((size_t)b * LL + st) * DM + c];
  }
  __shared__ float ls[DM];
  __shared__ float pr[DM];
  ls[c] = siluf_(acc);
  __syncthreads();
  const int j   = c & 31;                     // output index 0..31
  const int seg = c >> 5;                     // 8 partial segments of 32
  float p = 0.f;
  const float* wr = xpw + (size_t)j * DM + seg * 32;
  const float* er = ls + seg * 32;
#pragma unroll
  for (int q = 0; q < 32; ++q) p += wr[q] * er[q];
  pr[c] = p;
  __syncthreads();
  if (seg == 0) {
    float s = 0.f;
#pragma unroll
    for (int q = 0; q < 8; ++q) s += pr[q * 32 + j];
    dbl[(size_t)pos * 32 + j] = s;
  }
}

// ---------------- dt = softplus(dtr @ dtw.T + dtb) ----------------
__global__ void __launch_bounds__(256)
dtproj_softplus(const float* __restrict__ dbl, const float* __restrict__ dtw,
                const float* __restrict__ dtb, float* __restrict__ dt) {
  const long long idx = (long long)blockIdx.x * blockDim.x + threadIdx.x;
  if (idx >= BLD) return;
  const int c = (int)(idx & (DM - 1));
  const long long pos = idx >> 8;
  const float* r = dbl + (size_t)pos * 32;    // dbl[0:16] = dtr
  float acc = dtb[c];
#pragma unroll
  for (int q = 0; q < DTR; ++q) acc += r[q] * dtw[c * DTR + q];
  dt[idx] = softplusf_(acc);
}

// ---------------- chunked selective scan ----------------
// Phase A: per (b,chunk,d,n): cumulative decay product + local end state (h0 = 0)
__global__ void __launch_bounds__(256)
scan_phaseA(const float* __restrict__ u, const float* __restrict__ dt,
            const float* __restrict__ dbl, const float* __restrict__ A_log,
            float* __restrict__ carryA, float* __restrict__ carryH) {
  const long long idx = (long long)blockIdx.x * blockDim.x + threadIdx.x;
  if (idx >= (long long)BN * NC * DM * NS) return;
  const int n  = (int)(idx & 7);
  long long t1 = idx >> 3;
  const int dd = (int)(t1 & (DM - 1));
  long long t2 = t1 >> 8;
  const int ch = (int)(t2 & (NC - 1));
  const int b  = (int)(t2 >> 7);
  const float A = -__expf(A_log[dd * NS + n]);
  float aprod = 1.f, h = 0.f;
  const size_t base_t = (size_t)b * LL + (size_t)ch * CH;
  for (int s = 0; s < CH; ++s) {
    const size_t bt = base_t + s;
    const float dtv = dt[bt * DM + dd];
    const float uv  = u[bt * DM + dd];
    const float Bv  = dbl[bt * 32 + DTR + n];
    const float dA  = __expf(dtv * A);
    h = dA * h + dtv * Bv * uv;
    aprod *= dA;
  }
  carryA[idx] = aprod;
  carryH[idx] = h;
}

// Phase B: serial prefix over chunks per (b,d,n) -> incoming state per chunk
__global__ void __launch_bounds__(256)
scan_phaseB(const float* __restrict__ carryA, const float* __restrict__ carryH,
            float* __restrict__ hin) {
  const int idx = blockIdx.x * blockDim.x + threadIdx.x;
  if (idx >= BN * DM * NS) return;
  const int n  = idx & 7;
  const int dd = (idx >> 3) & (DM - 1);
  const int b  = idx >> 11;
  float h = 0.f;
  for (int ch = 0; ch < NC; ++ch) {
    const size_t ci = ((((size_t)b * NC + ch) * DM + dd) << 3) + n;
    hin[ci] = h;
    h = carryA[ci] * h + carryH[ci];
  }
}

// Phase C: replay chunk with incoming state, emit y (+ u*Dp). rev: add at flipped index.
__global__ void __launch_bounds__(256)
scan_phaseC(const float* __restrict__ u, const float* __restrict__ dt,
            const float* __restrict__ dbl, const float* __restrict__ A_log,
            const float* __restrict__ Dp, const float* __restrict__ hin,
            float* __restrict__ y, int rev) {
  const long long idx = (long long)blockIdx.x * blockDim.x + threadIdx.x;
  if (idx >= (long long)BN * NC * DM) return;
  const int dd = (int)(idx & (DM - 1));
  long long t  = idx >> 8;
  const int ch = (int)(t & (NC - 1));
  const int b  = (int)(t >> 7);
  float A[NS], h[NS];
  const size_t ci = (((size_t)b * NC + ch) * DM + dd) << 3;
#pragma unroll
  for (int n = 0; n < NS; ++n) { A[n] = -__expf(A_log[dd * NS + n]); h[n] = hin[ci + n]; }
  const float Dv = Dp[dd];
  const size_t base_t = (size_t)b * LL + (size_t)ch * CH;
  for (int s = 0; s < CH; ++s) {
    const size_t bt = base_t + s;
    const float dtv = dt[bt * DM + dd];
    const float uv  = u[bt * DM + dd];
    const float* bc = dbl + bt * 32;
    float acc = uv * Dv;
#pragma unroll
    for (int n = 0; n < NS; ++n) {
      const float dA = __expf(dtv * A[n]);
      h[n] = dA * h[n] + dtv * bc[DTR + n] * uv;
      acc += h[n] * bc[DTR + NS + n];
    }
    const int tt = ch * CH + s;
    if (rev) {
      const size_t oi = ((size_t)b * LL + (LL - 1 - tt)) * DM + dd;
      y[oi] += acc;
    } else {
      y[bt * DM + dd] = acc;
    }
  }
}

// ---------------- out-layernorm of 0.5*(y_f + flip(y_b)), gate with silu(z) -> bf16 ----
__global__ void __launch_bounds__(256)
outnorm_gate(const float* __restrict__ y, const float* __restrict__ z,
             const float* __restrict__ w, const float* __restrict__ b,
             __bf16* __restrict__ ybf) {
  const int row = blockIdx.x;
  const int c = threadIdx.x;
  const float v = 0.5f * y[(size_t)row * DM + c];
  __shared__ float red[DM];
  red[c] = v; __syncthreads();
  for (int s = 128; s > 0; s >>= 1) { if (c < s) red[c] += red[c + s]; __syncthreads(); }
  const float mu = red[0] * (1.0f / DM);
  __syncthreads();
  const float dv = v - mu;
  red[c] = dv * dv; __syncthreads();
  for (int s = 128; s > 0; s >>= 1) { if (c < s) red[c] += red[c + s]; __syncthreads(); }
  const float var = red[0] * (1.0f / DM);
  const float ln = dv * rsqrtf(var + 1e-5f) * w[c] + b[c];
  const float zz = z[(size_t)row * DM + c];
  ybf[(size_t)row * DM + c] = f2bf(ln * siluf_(zz));
}

// ---------------- host launcher ----------------
extern "C" void kernel_launch(void* const* d_in, const int* in_sizes, int n_in,
                              void* d_out, int out_size, void* d_ws, size_t ws_size,
                              hipStream_t stream) {
  const float* input0   = (const float*)d_in[0];
  const float* input1   = (const float*)d_in[1];
  const float* norm0_w  = (const float*)d_in[2];
  const float* norm0_b  = (const float*)d_in[3];
  const float* norm1_w  = (const float*)d_in[4];
  const float* norm1_b  = (const float*)d_in[5];
  const float* in_proj_w       = (const float*)d_in[6];
  const float* in_proj_extra_w = (const float*)d_in[7];
  const float* convx_w  = (const float*)d_in[8];
  const float* convx_b  = (const float*)d_in[9];
  const float* conve_w  = (const float*)d_in[10];
  const float* conve_b  = (const float*)d_in[11];
  const float* convx_w_r = (const float*)d_in[12];
  const float* convx_b_r = (const float*)d_in[13];
  const float* conve_w_r = (const float*)d_in[14];
  const float* conve_b_r = (const float*)d_in[15];
  const float* xproj_w   = (const float*)d_in[16];
  const float* xproj_w_r = (const float*)d_in[17];
  const float* dtproj_w  = (const float*)d_in[18];
  const float* dtproj_b  = (const float*)d_in[19];
  const float* dtproj_w_r = (const float*)d_in[20];
  const float* dtproj_b_r = (const float*)d_in[21];
  const float* A_log    = (const float*)d_in[22];
  const float* A_log_r  = (const float*)d_in[23];
  const float* Dp       = (const float*)d_in[24];
  const float* Dp_r     = (const float*)d_in[25];
  const float* outnorm_w = (const float*)d_in[26];
  const float* outnorm_b = (const float*)d_in[27];
  const float* outproj_w = (const float*)d_in[28];

  // scratch layout
  char* cur = (char*)d_ws;
  auto alloc = [&](size_t bytes) -> void* {
    void* r = (void*)cur;
    cur += (bytes + 255) & ~(size_t)255;
    return r;
  };
  __bf16* h0b   = (__bf16*)alloc(BLD * 2);
  __bf16* h1b   = (__bf16*)alloc(BLD * 2);
  __bf16* wInB  = (__bf16*)alloc((size_t)512 * 256 * 2);
  __bf16* wExB  = (__bf16*)alloc((size_t)256 * 256 * 2);
  __bf16* wOutB = (__bf16*)alloc((size_t)256 * 256 * 2);
  float* xbuf   = (float*)alloc(BLD * 4);
  float* zbuf   = (float*)alloc(BLD * 4);
  float* xebuf  = (float*)alloc(BLD * 4);
  float* xc_f   = (float*)alloc(BLD * 4);
  float* dt_f   = (float*)alloc(BLD * 4);
  float* dbl_f  = (float*)alloc((size_t)BL * 32 * 4);
  float* xc_b   = (float*)alloc(BLD * 4);
  float* dt_b   = (float*)alloc(BLD * 4);
  float* dbl_b  = (float*)alloc((size_t)BL * 32 * 4);
  float* ybuf   = (float*)alloc(BLD * 4);
  __bf16* ybf   = (__bf16*)alloc(BLD * 2);
  const size_t carryN = (size_t)BN * NC * DM * NS;
  float* carryA = (float*)alloc(carryN * 4);
  float* carryH = (float*)alloc(carryN * 4);
  float* hin    = (float*)alloc(carryN * 4);

  const dim3 thr256(256), thr128(128);

  // 1) input layernorms -> bf16 activations
  ln_to_bf16<<<BL, thr256, 0, stream>>>(input0, norm0_w, norm0_b, h0b);
  ln_to_bf16<<<BL, thr256, 0, stream>>>(input1, norm1_w, norm1_b, h1b);

  // 2) weights -> bf16
  cvt_f32_bf16<<<(512 * 256 + 255) / 256, thr256, 0, stream>>>(in_proj_w, wInB, 512 * 256);
  cvt_f32_bf16<<<(256 * 256 + 255) / 256, thr256, 0, stream>>>(in_proj_extra_w, wExB, 256 * 256);
  cvt_f32_bf16<<<(256 * 256 + 255) / 256, thr256, 0, stream>>>(outproj_w, wOutB, 256 * 256);

  // 3) xz = h0 @ in_proj_w.T (split x/z), xe = h1 @ in_proj_extra_w.T  [WMMA]
  gemm_bf16_wmma<<<(BL / 16) * (512 / 16) / 4, thr128, 0, stream>>>(
      h0b, wInB, BL, 512, 256, 1, xbuf, zbuf, nullptr);
  gemm_bf16_wmma<<<(BL / 16) * (256 / 16) / 4, thr128, 0, stream>>>(
      h1b, wExB, BL, 256, 256, 0, xebuf, nullptr, nullptr);

  const int blkBLD = (int)((BLD + 255) / 256);

  // 4) forward direction pre-scan elementwise pipeline
  conv_silu<<<blkBLD, thr256, 0, stream>>>(xbuf, convx_w, convx_b, xc_f, 0);
  conve_xproj<<<BL, thr256, 0, stream>>>(xebuf, conve_w, conve_b, xproj_w, dbl_f, 0);
  dtproj_softplus<<<blkBLD, thr256, 0, stream>>>(dbl_f, dtproj_w, dtproj_b, dt_f);

  // 5) backward direction (flipped reads)
  conv_silu<<<blkBLD, thr256, 0, stream>>>(xbuf, convx_w_r, convx_b_r, xc_b, 1);
  conve_xproj<<<BL, thr256, 0, stream>>>(xebuf, conve_w_r, conve_b_r, xproj_w_r, dbl_b, 1);
  dtproj_softplus<<<blkBLD, thr256, 0, stream>>>(dbl_b, dtproj_w_r, dtproj_b_r, dt_b);

  // 6) forward chunked selective scan -> y
  scan_phaseA<<<(int)(((long long)BN * NC * DM * NS + 255) / 256), thr256, 0, stream>>>(
      xc_f, dt_f, dbl_f, A_log, carryA, carryH);
  scan_phaseB<<<(BN * DM * NS + 255) / 256, thr256, 0, stream>>>(carryA, carryH, hin);
  scan_phaseC<<<(int)(((long long)BN * NC * DM + 255) / 256), thr256, 0, stream>>>(
      xc_f, dt_f, dbl_f, A_log, Dp, hin, ybuf, 0);

  // 7) backward chunked selective scan -> y += flip(y_b)
  scan_phaseA<<<(int)(((long long)BN * NC * DM * NS + 255) / 256), thr256, 0, stream>>>(
      xc_b, dt_b, dbl_b, A_log_r, carryA, carryH);
  scan_phaseB<<<(BN * DM * NS + 255) / 256, thr256, 0, stream>>>(carryA, carryH, hin);
  scan_phaseC<<<(int)(((long long)BN * NC * DM + 255) / 256), thr256, 0, stream>>>(
      xc_b, dt_b, dbl_b, A_log_r, Dp_r, hin, ybuf, 1);

  // 8) out layernorm of 0.5*y, gate by silu(z) -> bf16
  outnorm_gate<<<BL, thr256, 0, stream>>>(ybuf, zbuf, outnorm_w, outnorm_b, ybf);

  // 9) out = ybf @ outproj_w.T + skip   [WMMA]
  gemm_bf16_wmma<<<(BL / 16) * (256 / 16) / 4, thr128, 0, stream>>>(
      ybf, wOutB, BL, 256, 256, 2, (float*)d_out, nullptr, input0);
}
